// WindowAttention_16174846836952
// MI455X (gfx1250) — matrix operand order repaired and checked by reading, not compiled, not run on hardware
//
#include <hip/hip_runtime.h>

// ---------------------------------------------------------------------------
// Swin window attention, fused per-window, f16 WMMA (gfx1250 / CDNA5).
// DIM=384, HEADS=12, head_dim=32, N=49 tokens (padded to 64), B=4096 windows.
// ---------------------------------------------------------------------------

#define DIM    384
#define HEADS  12
#define HD     32
#define NTOK   49
#define NPAD   64
#define QK_SCALE 0.17677669529663687f   // 32^-0.5

typedef __attribute__((ext_vector_type(16))) _Float16       f16x16;
typedef __attribute__((ext_vector_type(8)))  float          f32x8;
typedef __attribute__((ext_vector_type(8)))  unsigned short u16x8;
typedef __attribute__((ext_vector_type(16))) unsigned short u16x16;

static __device__ __forceinline__ unsigned short f2h(float f) {
  return __builtin_bit_cast(unsigned short, (_Float16)f);
}

// A fragment (16x32 f16, M x K): lane holds row m = lane&15; K chunks of 8 at
// {8h, 16+8h} with h = lane>>4. Source is row-major [M][K], K contiguous.
static __device__ __forceinline__ f16x16 frag_a(const unsigned short* p, int stride) {
  const int lane = threadIdx.x & 31;
  const int m = lane & 15, h = lane >> 4;
  const unsigned short* r = p + m * stride;
  u16x8 lo = *(const u16x8*)(r + 8 * h);
  u16x8 hi = *(const u16x8*)(r + 16 + 8 * h);
  u16x16 f = __builtin_shufflevector(lo, hi, 0,1,2,3,4,5,6,7,8,9,10,11,12,13,14,15);
  return __builtin_bit_cast(f16x16, f);
}

// B fragment (32x16 f16, K x N): lane holds col n = lane&15; K chunks of 8 at
// {8h, 16+8h}. Source stored transposed, row-major [N][K].
static __device__ __forceinline__ f16x16 frag_b(const unsigned short* p, int stride) {
  const int lane = threadIdx.x & 31;
  const int n = lane & 15, h = lane >> 4;
  const unsigned short* r = p + n * stride;
  u16x8 lo = *(const u16x8*)(r + 8 * h);
  u16x8 hi = *(const u16x8*)(r + 16 + 8 * h);
  u16x16 f = __builtin_shufflevector(lo, hi, 0,1,2,3,4,5,6,7,8,9,10,11,12,13,14,15);
  return __builtin_bit_cast(f16x16, f);
}

static __device__ __forceinline__ f32x8 wmma_f16(f16x16 a, f16x16 b, f32x8 c) {
  return __builtin_amdgcn_wmma_f32_16x16x32_f16(false, a, false, b, (short)0, c,
                                                false, false);
}

// ---------------------------------------------------------------------------
// Kernel 0: convert Wq / Wkv / Wp (fp32, [K][N]) -> transposed f16 [N][K] in ws.
// ---------------------------------------------------------------------------
__global__ void prep_weights(const float* __restrict__ Wq,
                             const float* __restrict__ Wkv,
                             const float* __restrict__ Wp,
                             unsigned short* __restrict__ wT) {
  const int i = blockIdx.x * 256 + threadIdx.x;            // 589824 total
  if (i < 147456) {                                        // WqT [384][384]
    int n = i / 384, k = i % 384;
    wT[i] = f2h(Wq[k * 384 + n]);
  } else if (i < 147456 + 294912) {                        // WkvT [768][384]
    int j = i - 147456;
    int n = j / 384, k = j % 384;
    wT[i] = f2h(Wkv[k * 768 + n]);
  } else {                                                 // WpT [384][384]
    int j = i - 442368;
    int n = j / 384, k = j % 384;
    wT[i] = f2h(Wp[k * 384 + n]);
  }
}

// ---------------------------------------------------------------------------
// Kernel 1: one workgroup (8 wave32) per window. Fully fused attention block.
// LDS map (bytes), 268304 total:
//   [0      .. 49152)  sQin  (64x384 f16)    -- overlaid after projections by:
//   [49152  .. 98304)  sKVin (64x384 f16)        sS (64x64 f32)  @ 0
//                                                sP (64x64 f16)  @ 16384
//                                                sO (64x384 f16) @ 24576
//   [98304  ..147456)  sQp   (64x384 f16, pre-scaled)
//   [147456 ..196608)  sK    (64x384 f16)
//   [196608 ..245760)  sVt   (384x64 f16, V transposed per head-dim)
//   [245760 ..250576)  relS  (49x49 u16, rel_index*HEADS)
//   [250576 ..260192)  maskS (49x49 f32, this window's mask slice)
//   [260192 ..268304)  biasT (169x12 f32, bias_table)
// ---------------------------------------------------------------------------
__global__ __launch_bounds__(256, 1)
void window_attn(const float* __restrict__ q,
                 const float* __restrict__ kv,
                 const float* __restrict__ mask,
                 const float* __restrict__ bq,
                 const float* __restrict__ bkv,
                 const float* __restrict__ bp,
                 const float* __restrict__ bias_table,
                 const int*   __restrict__ rel_index,
                 const unsigned short* __restrict__ wT,
                 float* __restrict__ out) {
  __shared__ __attribute__((aligned(16))) unsigned char smem[268304];

  unsigned short* sQin  = (unsigned short*)(smem);
  unsigned short* sKVin = (unsigned short*)(smem + 49152);
  float*          sS    = (float*)(smem);
  unsigned short* sP    = (unsigned short*)(smem + 16384);
  unsigned short* sO    = (unsigned short*)(smem + 24576);
  unsigned short* sQp   = (unsigned short*)(smem + 98304);
  unsigned short* sK    = (unsigned short*)(smem + 147456);
  unsigned short* sVt   = (unsigned short*)(smem + 196608);
  unsigned short* relS  = (unsigned short*)(smem + 245760);
  float*          maskS = (float*)(smem + 250576);
  float*          biasT = (float*)(smem + 260192);

  const unsigned short* wqT  = wT;
  const unsigned short* wkvT = wT + 147456;
  const unsigned short* wpT  = wT + 442368;

  const int b    = blockIdx.x;
  const int w    = b & 63;         // mask window index
  const int tid  = threadIdx.x;
  const int wave = tid >> 5;
  const int lane = tid & 31;
  const int fn   = lane & 15;      // fragment col (B/C/D) or row (A)
  const int fh   = lane >> 4;

  // ---- Phase 0: stage q / kv (f16) + rel/mask/bias tables -----------------
  {
    const float4* qv  = (const float4*)(q  + (size_t)b * NTOK * DIM);
    const float4* kvv = (const float4*)(kv + (size_t)b * NTOK * DIM);
    for (int e = tid; e < NPAD * (DIM / 4); e += 256) {     // 6144 float4s
      int row = e / (DIM / 4);
      int c4  = e % (DIM / 4);
      float4 a = (row < NTOK) ? qv[row * (DIM / 4) + c4] : make_float4(0, 0, 0, 0);
      float4 c = (row < NTOK) ? kvv[row * (DIM / 4) + c4] : make_float4(0, 0, 0, 0);
      int o = row * DIM + c4 * 4;
      sQin[o + 0] = f2h(a.x); sQin[o + 1] = f2h(a.y);
      sQin[o + 2] = f2h(a.z); sQin[o + 3] = f2h(a.w);
      sKVin[o + 0] = f2h(c.x); sKVin[o + 1] = f2h(c.y);
      sKVin[o + 2] = f2h(c.z); sKVin[o + 3] = f2h(c.w);
    }
    for (int i = tid; i < NTOK * NTOK; i += 256) {          // 2401
      relS[i]  = (unsigned short)(rel_index[i] * HEADS);
      maskS[i] = mask[(size_t)w * NTOK * NTOK + i];
    }
    for (int i = tid; i < 169 * HEADS; i += 256)            // 2028
      biasT[i] = bias_table[i];
  }
  __syncthreads();

  // ---- Phase 1: projections qp = q@Wq (scaled), [k|v] = kv@Wkv ------------
  // Units of (row-tile, group of 3 col-tiles): qp 4x8, kv 4x16 -> 96 units.
  for (int u = wave; u < 96; u += 8) {
    const bool isQ = (u < 32);
    const int uu = isQ ? u : (u - 32);
    const int groups = isQ ? 8 : 16;
    const int rt = uu / groups, cg = uu % groups;
    const unsigned short* src = isQ ? sQin : sKVin;
    const unsigned short* wgt = isQ ? wqT : wkvT;
    f32x8 acc[3];
    acc[0] = acc[1] = acc[2] = (f32x8){0.f,0.f,0.f,0.f,0.f,0.f,0.f,0.f};
    for (int ks = 0; ks < 12; ks++) {
      f16x16 a = frag_a(src + rt * 16 * DIM + ks * 32, DIM);
#pragma unroll
      for (int j = 0; j < 3; j++) {
        f16x16 bb = frag_b(wgt + (cg * 48 + j * 16) * DIM + ks * 32, DIM);
        acc[j] = wmma_f16(a, bb, acc[j]);
      }
    }
#pragma unroll
    for (int j = 0; j < 3; j++) {
      const int colg = cg * 48 + j * 16 + fn;
      if (isQ) {
        const float bias = bq[colg];
        for (int r = 0; r < 8; r++) {
          int row = rt * 16 + 8 * fh + r;
          sQp[row * DIM + colg] = f2h((acc[j][r] + bias) * QK_SCALE);
        }
      } else {
        const float bias = bkv[colg];
        if (colg < DIM) {                     // K half
          for (int r = 0; r < 8; r++) {
            int row = rt * 16 + 8 * fh + r;
            sK[row * DIM + colg] = f2h(acc[j][r] + bias);
          }
        } else {                              // V half -> transposed store
          int c2 = colg - DIM;
          for (int r = 0; r < 8; r++) {
            int row = rt * 16 + 8 * fh + r;
            sVt[c2 * NPAD + row] = f2h(acc[j][r] + bias);
          }
        }
      }
    }
  }
  __syncthreads();

  // ---- Phase 2: per-head attention ---------------------------------------
  for (int h = 0; h < HEADS; h++) {
    // (a) S = Q_h * K_h^T (+ rel-pos bias + window mask), 4x4 tiles
    for (int t = wave; t < 16; t += 8) {
      const int mt = t >> 2, nt = t & 3;
      f32x8 acc = {0.f, 0.f, 0.f, 0.f, 0.f, 0.f, 0.f, 0.f};
      f16x16 a  = frag_a(sQp + mt * 16 * DIM + h * HD, DIM);
      f16x16 bb = frag_b(sK  + nt * 16 * DIM + h * HD, DIM);
      acc = wmma_f16(a, bb, acc);
      const int col = nt * 16 + fn;
      for (int r = 0; r < 8; r++) {
        const int row = mt * 16 + 8 * fh + r;
        float add = 0.f;
        if (row < NTOK && col < NTOK) {
          const int ij = row * NTOK + col;
          add = biasT[relS[ij] + h] + maskS[ij];
        }
        sS[row * NPAD + col] = acc[r] + add;
      }
    }
    __syncthreads();

    // (b) softmax: 4 threads per row, shfl reductions, emit normalized f16 P
    {
      const int row = tid >> 2, sub = tid & 3;
      const int c0 = sub * 16;
      const bool valid = row < NTOK;
      float mx = -3.0e38f;
      if (valid)
        for (int c = c0; c < c0 + 16 && c < NTOK; c++)
          mx = fmaxf(mx, sS[row * NPAD + c]);
      mx = fmaxf(mx, __shfl_xor(mx, 1, 4));
      mx = fmaxf(mx, __shfl_xor(mx, 2, 4));
      float sum = 0.f;
      if (valid)
        for (int c = c0; c < c0 + 16 && c < NTOK; c++) {
          float e = __expf(sS[row * NPAD + c] - mx);
          sum += e;
          sS[row * NPAD + c] = e;
        }
      sum += __shfl_xor(sum, 1, 4);
      sum += __shfl_xor(sum, 2, 4);
      const float inv = 1.0f / sum;
      for (int c = c0; c < c0 + 16; c++) {
        unsigned short v = 0;
        if (valid && c < NTOK) v = f2h(sS[row * NPAD + c] * inv);
        sP[row * NPAD + c] = v;
      }
    }
    __syncthreads();

    // (c) O_h = P * V_h : 4x2 tiles, 2 K-steps (K = 64 padded tokens)
    {
      const int mt = wave >> 1, nt = wave & 1;
      f32x8 acc = {0.f, 0.f, 0.f, 0.f, 0.f, 0.f, 0.f, 0.f};
      for (int ks = 0; ks < 2; ks++) {
        f16x16 a  = frag_a(sP + mt * 16 * NPAD + ks * 32, NPAD);
        f16x16 bb = frag_b(sVt + (h * HD + nt * 16) * NPAD + ks * 32, NPAD);
        acc = wmma_f16(a, bb, acc);
      }
      for (int r = 0; r < 8; r++)
        sO[(mt * 16 + 8 * fh + r) * DIM + h * HD + nt * 16 + fn] = f2h(acc[r]);
    }
    __syncthreads();
  }

  // ---- Phase 3: final projection out = O @ Wp + bp ------------------------
  // Units of (row-tile, group of 3 col-tiles): 4x8 = 32 units.
  for (int u = wave; u < 32; u += 8) {
    const int rt = u / 8, cg = u % 8;
    f32x8 acc[3];
    acc[0] = acc[1] = acc[2] = (f32x8){0.f,0.f,0.f,0.f,0.f,0.f,0.f,0.f};
    for (int ks = 0; ks < 12; ks++) {
      f16x16 a = frag_a(sO + rt * 16 * DIM + ks * 32, DIM);
#pragma unroll
      for (int j = 0; j < 3; j++) {
        f16x16 bb = frag_b(wpT + (cg * 48 + j * 16) * DIM + ks * 32, DIM);
        acc[j] = wmma_f16(a, bb, acc[j]);
      }
    }
#pragma unroll
    for (int j = 0; j < 3; j++) {
      const int col = cg * 48 + j * 16 + fn;
      const float bias = bp[col];
      for (int r = 0; r < 8; r++) {
        int row = rt * 16 + 8 * fh + r;
        if (row < NTOK)
          out[((size_t)b * NTOK + row) * DIM + col] = acc[j][r] + bias;
      }
    }
  }
}

// ---------------------------------------------------------------------------
extern "C" void kernel_launch(void* const* d_in, const int* in_sizes, int n_in,
                              void* d_out, int out_size, void* d_ws, size_t ws_size,
                              hipStream_t stream) {
  const float* q          = (const float*)d_in[0];
  const float* kv         = (const float*)d_in[1];
  const float* mask       = (const float*)d_in[2];
  const float* Wq         = (const float*)d_in[3];
  const float* bq         = (const float*)d_in[4];
  const float* Wkv        = (const float*)d_in[5];
  const float* bkv        = (const float*)d_in[6];
  const float* Wp         = (const float*)d_in[7];
  const float* bp         = (const float*)d_in[8];
  const float* bias_table = (const float*)d_in[9];
  const int*   rel_index  = (const int*)d_in[10];

  unsigned short* wT = (unsigned short*)d_ws;   // 589824 f16 = ~1.13 MB

  prep_weights<<<2304, 256, 0, stream>>>(Wq, Wkv, Wp, wT);
  window_attn<<<4096, 256, 0, stream>>>(q, kv, mask, bq, bkv, bp,
                                        bias_table, rel_index, wT,
                                        (float*)d_out);
}
